// NPUDeepseekMLAAttention_8280696947358
// MI455X (gfx1250) — matrix operand order repaired and checked by reading, not compiled, not run on hardware
//
#include <hip/hip_runtime.h>
#include <hip/hip_bf16.h>

// ---------------------------------------------------------------------------
// DeepSeek MLA decode for gfx1250 (MI455X).
//  - Projections (weight-streaming, memory-bound): fp32 V_WMMA_F32_16X16X4_F32
//  - Attention (compute-leaning, 128 heads share keys): bf16-in/f32-acc
//    V_WMMA_F32_16X16X32_BF16 (8x MAC depth per instruction)
// ---------------------------------------------------------------------------

typedef __attribute__((ext_vector_type(2)))  float  v2f;
typedef __attribute__((ext_vector_type(8)))  float  v8f;
typedef __attribute__((ext_vector_type(16))) __bf16 v16bf;

#define B_    32
#define HID_  5120
#define H_    128
#define DN_   128
#define DR_   64
#define DV_   128
#define DQH_  192            // DN + DR per head
#define QLR_  1536
#define KVLR_ 512
#define DK_   576            // KVLR + DR
#define BLK_  128
#define NBPS_ 16
#define SQ_   2048           // NBPS * BLK

__device__ __forceinline__ v8f wmma4(v2f a, v2f b, v8f c) {
  // D = A(16x4) * B(4x16) + C, fp32, wave32
  return __builtin_amdgcn_wmma_f32_16x16x4_f32(false, a, false, b, (short)0, c,
                                               false, false);
}

__device__ __forceinline__ v8f wmma32bf(v16bf a, v16bf b, v8f c) {
  // D = A(16x32) * B(32x16) + C, bf16 in / f32 acc, wave32
  return __builtin_amdgcn_wmma_f32_16x16x32_bf16(false, a, false, b, (short)0,
                                                 c, false, false);
}

// A fragment (16x32, contiguous k order shared with B): element j -> K = 2*kbase+j;
// kbase = 0 (lanes 0-15) or 8 (lanes 16-31); 16 contiguous values.
__device__ __forceinline__ v16bf cvt_a16(const float* __restrict__ p,
                                         int kbase) {
  v16bf r;
#pragma unroll
  for (int j = 0; j < 8; ++j) {
    r[j]     = (__bf16)p[2 * kbase + j];
    r[j + 8] = (__bf16)p[2 * kbase + 8 + j];
  }
  return r;
}

// B fragment (32x16, contiguous k order shared with A): element j -> K = boff+j;
// boff = 0 (lanes 0-15) or 16 (lanes 16-31); 16 contiguous values.
__device__ __forceinline__ v16bf cvt_b16(const float* __restrict__ p,
                                         int boff) {
  v16bf r;
#pragma unroll
  for (int j = 0; j < 16; ++j) r[j] = (__bf16)p[boff + j];
  return r;
}

// ---------------------------------------------------------------------------
// Generic strided-batched fp32 GEMM: C[z] = A[z](MxK, lda) * B[z](KxN, ldb)
// one wave per 16x64 output strip, K stepped by 4 through the WMMA unit.
// Requires M%16==0, N%64==0, K%4==0 (true for every use here).
// ---------------------------------------------------------------------------
__global__ __launch_bounds__(32)
void gemm_wmma_f32(const float* __restrict__ A, long sA, int lda,
                   const float* __restrict__ B, long sB, int ldb,
                   float* __restrict__ C, long sC, int ldc,
                   int K) {
  const int lane = threadIdx.x;
  const int n0 = blockIdx.x * 64;
  const int m0 = blockIdx.y * 16;
  const int z  = blockIdx.z;
  const int i    = lane & 15;
  const int koff = (lane < 16) ? 0 : 2;

  const float* ap = A + (long)z * sA + (long)(m0 + i) * lda + koff;
  const float* bp = B + (long)z * sB + (long)koff * ldb + n0 + i;
  const long ldb4 = 4L * (long)ldb;
  v8f c0 = {}, c1 = {}, c2 = {}, c3 = {};
  for (int k0 = 0; k0 < K; k0 += 4) {
    v2f a;
    a.x = ap[0]; a.y = ap[1]; ap += 4;
    const float* bq = bp + ldb;
    v2f b0, b1, b2, b3;
    b0.x = bp[0];  b1.x = bp[16]; b2.x = bp[32]; b3.x = bp[48];
    b0.y = bq[0];  b1.y = bq[16]; b2.y = bq[32]; b3.y = bq[48];
    __builtin_prefetch(bp + 32L * (long)ldb, 0, 1);  // stream B ahead
    bp += ldb4;
    c0 = wmma4(a, b0, c0);
    c1 = wmma4(a, b1, c1);
    c2 = wmma4(a, b2, c2);
    c3 = wmma4(a, b3, c3);
  }
  float* crow = C + (long)z * sC + n0 + i;
  const int rbase = (lane < 16) ? 0 : 8;
  for (int r = 0; r < 8; ++r) {
    float* cr = crow + (long)(m0 + rbase + r) * ldc;
    cr[0] = c0[r]; cr[16] = c1[r]; cr[32] = c2[r]; cr[48] = c3[r];
  }
}

// ---------------------------------------------------------------------------
// RMSNorm over rows of length N (one block per row)
// ---------------------------------------------------------------------------
__global__ __launch_bounds__(256)
void rmsnorm_k(const float* __restrict__ x, const float* __restrict__ w,
               float* __restrict__ y, int N) {
  __shared__ float red[256];
  const int b = blockIdx.x, t = threadIdx.x;
  const float* xr = x + (long)b * N;
  float* yr = y + (long)b * N;
  float s = 0.f;
  for (int j = t; j < N; j += 256) { float v = xr[j]; s += v * v; }
  red[t] = s; __syncthreads();
  for (int off = 128; off > 0; off >>= 1) {
    if (t < off) red[t] += red[t + off];
    __syncthreads();
  }
  const float inv = rsqrtf(red[0] / (float)N + 1e-6f);
  for (int j = t; j < N; j += 256) yr[j] = xr[j] * inv * w[j];
}

// ---------------------------------------------------------------------------
// kv prep: rmsnorm(kv[:, :512]) -> kv_lat_new ; rope(kv[:, 512:576]) -> pe_new
// ---------------------------------------------------------------------------
__global__ __launch_bounds__(256)
void kv_prep(const float* __restrict__ kv, const float* __restrict__ w,
             const float* __restrict__ cosb, const float* __restrict__ sinb,
             float* __restrict__ kv_lat_new, float* __restrict__ pe_new) {
  __shared__ float red[256];
  const int b = blockIdx.x, t = threadIdx.x;
  const float* kr = kv + (long)b * DK_;
  float s = 0.f;
  for (int j = t; j < KVLR_; j += 256) { float v = kr[j]; s += v * v; }
  red[t] = s; __syncthreads();
  for (int off = 128; off > 0; off >>= 1) {
    if (t < off) red[t] += red[t + off];
    __syncthreads();
  }
  const float inv = rsqrtf(red[0] / (float)KVLR_ + 1e-6f);
  for (int j = t; j < KVLR_; j += 256)
    kv_lat_new[(long)b * KVLR_ + j] = kr[j] * inv * w[j];
  if (t < DR_ / 2) {
    const float x1 = kr[KVLR_ + 2 * t];
    const float x2 = kr[KVLR_ + 2 * t + 1];
    const float cc = cosb[b * (DR_ / 2) + t];
    const float ss = sinb[b * (DR_ / 2) + t];
    pe_new[(long)b * DR_ + 2 * t]     = x1 * cc - x2 * ss;
    pe_new[(long)b * DR_ + 2 * t + 1] = x2 * cc + x1 * ss;
  }
}

// ---------------------------------------------------------------------------
// In-place interleaved RoPE of the q_pe slice of q[B][H][192]
// ---------------------------------------------------------------------------
__global__ __launch_bounds__(32)
void rope_q(float* __restrict__ q, const float* __restrict__ cosb,
            const float* __restrict__ sinb) {
  const int b = blockIdx.x, h = blockIdx.y, t = threadIdx.x;  // t < 32
  float* p = q + (long)b * H_ * DQH_ + (long)h * DQH_ + DN_;
  const float x1 = p[2 * t], x2 = p[2 * t + 1];
  const float cc = cosb[b * (DR_ / 2) + t];
  const float ss = sinb[b * (DR_ / 2) + t];
  p[2 * t]     = x1 * cc - x2 * ss;
  p[2 * t + 1] = x2 * cc + x1 * ss;
}

// ---------------------------------------------------------------------------
// scores[b][h][s] = (q_lat . key_lat + q_pe . key_pe) * scale
// bf16 WMMA, 16h x 64s strip per wave, K = 576 stepped by 32.
// Keys gathered through block_table with the slot_mapping override.
// ---------------------------------------------------------------------------
__global__ __launch_bounds__(32)
void mla_scores(const float* __restrict__ qlat, const float* __restrict__ qbuf,
                const float* __restrict__ cache_kv,
                const float* __restrict__ cache_pe,
                const float* __restrict__ kv_new,
                const float* __restrict__ pe_new,
                const int* __restrict__ block_table,
                const int* __restrict__ slot_mapping,
                const int* __restrict__ seq_lens,
                float* __restrict__ scores, float scale) {
  const int lane = threadIdx.x;
  const int s0 = blockIdx.x * 64;
  const int h0 = blockIdx.y * 16;
  const int b  = blockIdx.z;
  const int i     = lane & 15;
  const int kbase = (lane < 16) ? 0 : 8;
  const int boff  = (lane < 16) ? 0 : 16;

  // A: per-lane fixed head row (q_lat 512 then q_pe 64)
  const int head = h0 + i;
  const float* pa  = qlat + ((long)b * H_ + head) * KVLR_;
  const float* pae = qbuf + (long)b * H_ * DQH_ + (long)head * DQH_ + DN_;

  // B: per-lane fixed sequence columns (4 tiles), gathered via block_table
  const int slot = slot_mapping[b];
  const int* bt  = block_table + b * NBPS_;
  const int sA = s0 + i, sB = s0 + 16 + i, sC = s0 + 32 + i, sD = s0 + 48 + i;
  const float* kvn = kv_new + (long)b * KVLR_;
  const float* pen = pe_new + (long)b * DR_;
#define GATHER(sv, pk, pp)                                                   \
  const float *pk, *pp;                                                      \
  {                                                                          \
    const int flat = bt[(sv) >> 7] * BLK_ + ((sv) & (BLK_ - 1));             \
    const bool ovr = (flat == slot);                                         \
    pk = ovr ? kvn : cache_kv + (long)flat * KVLR_;                          \
    pp = ovr ? pen : cache_pe + (long)flat * DR_;                            \
  }
  GATHER(sA, pk0, pp0)
  GATHER(sB, pk1, pp1)
  GATHER(sC, pk2, pp2)
  GATHER(sD, pk3, pp3)
#undef GATHER

  v8f c0 = {}, c1 = {}, c2 = {}, c3 = {};
  for (int k0 = 0; k0 < KVLR_; k0 += 32) {
    const v16bf a = cvt_a16(pa + k0, kbase);
    c0 = wmma32bf(a, cvt_b16(pk0 + k0, boff), c0);
    c1 = wmma32bf(a, cvt_b16(pk1 + k0, boff), c1);
    c2 = wmma32bf(a, cvt_b16(pk2 + k0, boff), c2);
    c3 = wmma32bf(a, cvt_b16(pk3 + k0, boff), c3);
  }
  for (int k0 = 0; k0 < DR_; k0 += 32) {
    const v16bf a = cvt_a16(pae + k0, kbase);
    c0 = wmma32bf(a, cvt_b16(pp0 + k0, boff), c0);
    c1 = wmma32bf(a, cvt_b16(pp1 + k0, boff), c1);
    c2 = wmma32bf(a, cvt_b16(pp2 + k0, boff), c2);
    c3 = wmma32bf(a, cvt_b16(pp3 + k0, boff), c3);
  }

  const int sl = seq_lens[b];
  const int rbase = (lane < 16) ? 0 : 8;
  float* outb = scores + (long)b * H_ * SQ_;
  for (int r = 0; r < 8; ++r) {
    float* orow = outb + (long)(h0 + rbase + r) * SQ_;
    orow[sA] = (sA < sl) ? c0[r] * scale : -3.0e38f;
    orow[sB] = (sB < sl) ? c1[r] * scale : -3.0e38f;
    orow[sC] = (sC < sl) ? c2[r] * scale : -3.0e38f;
    orow[sD] = (sD < sl) ? c3[r] * scale : -3.0e38f;
  }
}

// ---------------------------------------------------------------------------
// Row softmax over S = 2048 (one block per (b,h) row)
// ---------------------------------------------------------------------------
__global__ __launch_bounds__(256)
void softmax_rows(float* __restrict__ x) {
  __shared__ float red[256];
  const int t = threadIdx.x;
  float* r = x + (long)blockIdx.x * SQ_;
  float m = -3.0e38f;
  for (int j = t; j < SQ_; j += 256) m = fmaxf(m, r[j]);
  red[t] = m; __syncthreads();
  for (int off = 128; off > 0; off >>= 1) {
    if (t < off) red[t] = fmaxf(red[t], red[t + off]);
    __syncthreads();
  }
  m = red[0]; __syncthreads();
  float s = 0.f;
  for (int j = t; j < SQ_; j += 256) {
    const float e = expf(r[j] - m);
    r[j] = e; s += e;
  }
  red[t] = s; __syncthreads();
  for (int off = 128; off > 0; off >>= 1) {
    if (t < off) red[t] += red[t + off];
    __syncthreads();
  }
  const float inv = 1.f / red[0];
  for (int j = t; j < SQ_; j += 256) r[j] *= inv;
}

// ---------------------------------------------------------------------------
// ctx[b][h][k] = sum_s attn[b][h][s] * key_lat[b][s][k]
// bf16 WMMA, 16h x 64k strip per wave, K = S = 2048 stepped by 32 with
// gathered B rows (block_table + slot_mapping override).
// ---------------------------------------------------------------------------
__global__ __launch_bounds__(32)
void mla_ctx(const float* __restrict__ attn,
             const float* __restrict__ cache_kv,
             const float* __restrict__ kv_new,
             const int* __restrict__ block_table,
             const int* __restrict__ slot_mapping,
             float* __restrict__ ctx) {
  const int lane = threadIdx.x;
  const int n0 = blockIdx.x * 64;
  const int h0 = blockIdx.y * 16;
  const int b  = blockIdx.z;
  const int i     = lane & 15;
  const int kbase = (lane < 16) ? 0 : 8;
  const int boff  = (lane < 16) ? 0 : 16;

  const int head = h0 + i;
  const float* arow = attn + ((long)b * H_ + head) * SQ_;
  const int* bt = block_table + b * NBPS_;
  const int slot = slot_mapping[b];
  const float* kvn = kv_new + (long)b * KVLR_ + n0 + i;

  v8f c0 = {}, c1 = {}, c2 = {}, c3 = {};
  for (int s0 = 0; s0 < SQ_; s0 += 32) {
    const v16bf a = cvt_a16(arow + s0, kbase);
    v16bf b0, b1, b2, b3;
#pragma unroll
    for (int j = 0; j < 16; ++j) {
      const int s = s0 + boff + j;
      const int f = bt[s >> 7] * BLK_ + (s & (BLK_ - 1));
      const float* kr =
          (f == slot) ? kvn : cache_kv + (long)f * KVLR_ + n0 + i;
      b0[j] = (__bf16)kr[0];
      b1[j] = (__bf16)kr[16];
      b2[j] = (__bf16)kr[32];
      b3[j] = (__bf16)kr[48];
    }
    __builtin_prefetch(cache_kv + ((long)(bt[s0 >> 7] * BLK_) + s0 + 64) * KVLR_
                           + n0 + i, 0, 1);  // stream keys ahead
    c0 = wmma32bf(a, b0, c0);
    c1 = wmma32bf(a, b1, c1);
    c2 = wmma32bf(a, b2, c2);
    c3 = wmma32bf(a, b3, c3);
  }
  const int rbase = (lane < 16) ? 0 : 8;
  for (int r = 0; r < 8; ++r) {
    float* cr = ctx + ((long)b * H_ + h0 + rbase + r) * KVLR_ + n0 + i;
    cr[0] = c0[r]; cr[16] = c1[r]; cr[32] = c2[r]; cr[48] = c3[r];
  }
}

// ---------------------------------------------------------------------------
// launch
// ---------------------------------------------------------------------------
extern "C" void kernel_launch(void* const* d_in, const int* in_sizes, int n_in,
                              void* d_out, int out_size, void* d_ws,
                              size_t ws_size, hipStream_t stream) {
  (void)in_sizes; (void)n_in; (void)out_size; (void)ws_size;
  const float* hidden   = (const float*)d_in[0];
  const float* cosb     = (const float*)d_in[1];
  const float* sinb     = (const float*)d_in[2];
  const float* Wq_a     = (const float*)d_in[3];
  const float* q_a_ln_w = (const float*)d_in[4];
  const float* Wq_b     = (const float*)d_in[5];
  const float* Wkv_a    = (const float*)d_in[6];
  const float* kv_a_ln_w= (const float*)d_in[7];
  const float* W_UK_T   = (const float*)d_in[8];
  const float* W_UV     = (const float*)d_in[9];
  const float* Wo       = (const float*)d_in[10];
  const float* cache_kv = (const float*)d_in[11];
  const float* cache_pe = (const float*)d_in[12];
  const int*   blk_tab  = (const int*)d_in[13];
  const int*   slot_map = (const int*)d_in[14];
  const int*   seq_lens = (const int*)d_in[15];
  float* out = (float*)d_out;

  // workspace carve-up (floats)
  float* ws = (float*)d_ws;
  size_t o = 0;
  float* qa     = ws + o; o += (size_t)B_ * QLR_;          // 32x1536
  float* qan    = ws + o; o += (size_t)B_ * QLR_;
  float* kv     = ws + o; o += (size_t)B_ * DK_;           // 32x576
  float* kvnew  = ws + o; o += (size_t)B_ * KVLR_;
  float* penew  = ws + o; o += (size_t)B_ * DR_;
  float* q      = ws + o; o += (size_t)B_ * H_ * DQH_;     // 32x128x192
  float* qlat   = ws + o; o += (size_t)B_ * H_ * KVLR_;    // 32x128x512
  float* attn   = ws + o; o += (size_t)B_ * H_ * SQ_;      // 32x128x2048
  float* ctx    = ws + o; o += (size_t)B_ * H_ * KVLR_;    // 32x128x512
  float* aout   = ws + o; o += (size_t)B_ * H_ * DV_;      // 32x128x128

  const float scale = 1.0f / sqrtf((float)(DN_ + DR_));

  // 1) q_a = hidden @ Wq_a     (32x5120 @ 5120x1536)
  gemm_wmma_f32<<<dim3(QLR_ / 64, B_ / 16, 1), 32, 0, stream>>>(
      hidden, 0, HID_, Wq_a, 0, QLR_, qa, 0, QLR_, HID_);
  // 2) kv = hidden @ Wkv_a     (32x5120 @ 5120x576)
  gemm_wmma_f32<<<dim3(DK_ / 64, B_ / 16, 1), 32, 0, stream>>>(
      hidden, 0, HID_, Wkv_a, 0, DK_, kv, 0, DK_, HID_);
  // 3) rmsnorm q_a
  rmsnorm_k<<<B_, 256, 0, stream>>>(qa, q_a_ln_w, qan, QLR_);
  // 4) kv latent rmsnorm + rope k_pe
  kv_prep<<<B_, 256, 0, stream>>>(kv, kv_a_ln_w, cosb, sinb, kvnew, penew);
  // 5) q = qan @ Wq_b          (32x1536 @ 1536x24576)
  gemm_wmma_f32<<<dim3((H_ * DQH_) / 64, B_ / 16, 1), 32, 0, stream>>>(
      qan, 0, QLR_, Wq_b, 0, H_ * DQH_, q, 0, H_ * DQH_, QLR_);
  // 6) rope q_pe in place
  rope_q<<<dim3(B_, H_), 32, 0, stream>>>(q, cosb, sinb);
  // 7) q_lat[b,h,:] = q_nope[b,h,:] @ W_UK_T[h]   (batch over h)
  gemm_wmma_f32<<<dim3(KVLR_ / 64, B_ / 16, H_), 32, 0, stream>>>(
      q,      DQH_,          H_ * DQH_,    // A: +h*192, lda=24576, K=128
      W_UK_T, DN_ * KVLR_,   KVLR_,        // B: +h*65536, ldb=512
      qlat,   KVLR_,         H_ * KVLR_,   // C: +h*512, ldc=65536
      DN_);
  // 8) scores (bf16 WMMA)
  mla_scores<<<dim3(SQ_ / 64, H_ / 16, B_), 32, 0, stream>>>(
      qlat, q, cache_kv, cache_pe, kvnew, penew, blk_tab, slot_map, seq_lens,
      attn, scale);
  // 9) softmax per (b,h)
  softmax_rows<<<B_ * H_, 256, 0, stream>>>(attn);
  // 10) ctx = attn @ keys (bf16 WMMA)
  mla_ctx<<<dim3(KVLR_ / 64, H_ / 16, B_), 32, 0, stream>>>(
      attn, cache_kv, kvnew, blk_tab, slot_map, ctx);
  // 11) aout[b,h,:] = ctx[b,h,:] @ W_UV[h]        (batch over h)
  gemm_wmma_f32<<<dim3(DV_ / 64, B_ / 16, H_), 32, 0, stream>>>(
      ctx,  KVLR_,        H_ * KVLR_,      // A: +h*512, lda=65536, K=512
      W_UV, KVLR_ * DV_,  DV_,             // B: +h*65536, ldb=128
      aout, DV_,          H_ * DV_,        // C: +h*128, ldc=16384
      KVLR_);
  // 12) out = aout @ Wo        (32x16384 @ 16384x5120)
  gemm_wmma_f32<<<dim3(HID_ / 64, B_ / 16, 1), 32, 0, stream>>>(
      aout, 0, H_ * DV_, Wo, 0, HID_, out, 0, HID_, H_ * DV_);
}
